// CGTGW_25615184953611
// MI455X (gfx1250) — compile-verified, matrix operand order checked by reading, not run on hardware
//
#include <hip/hip_runtime.h>
#include <hip/hip_bf16.h>
#include <math.h>

// ---------------------------------------------------------------------------
// HAKORN fused hyperbolic-NCA step for MI455X (gfx1250, wave32, WMMA).
// Two streaming passes over the N x N domain per step; gram tiles produced by
// v_wmma_f32_16x16x32_f16 (K=32 pads the 17-dim Minkowski contraction).
// ---------------------------------------------------------------------------

#define NN   4096
#define DP1  17
#define PADK 32
#define EPS_F 1e-7f

typedef __attribute__((ext_vector_type(16))) _Float16 v16h;
typedef __attribute__((ext_vector_type(8)))  _Float16 v8h;
typedef __attribute__((ext_vector_type(8)))  float    v8f;

// Load a 16x32 f16 operand tile in the CDNA5 WMMA A/B register layout.
// lane L<16 : row L, K=0..7 (elems 0-7) and K=16..23 (elems 8-15)
// lane L+16 : row L, K=8..15 and K=24..31
__device__ __forceinline__ v16h load_op16x32(const _Float16* tile, int stride_h, int lane) {
  const _Float16* p = tile + (size_t)(lane & 15) * stride_h + ((lane >> 4) << 3);
  v8h lo = *(const v8h*)p;
  v8h hi = *(const v8h*)(p + 16);
  v16h r;
#pragma unroll
  for (int e = 0; e < 8; ++e) { r[e] = lo[e]; r[e + 8] = hi[e]; }
  return r;
}

// sum across the 16 lanes of each wave32 half (C/D layout column lanes)
__device__ __forceinline__ float half16_reduce(float v) {
  v += __shfl_xor(v, 1, 32);
  v += __shfl_xor(v, 2, 32);
  v += __shfl_xor(v, 4, 32);
  v += __shfl_xor(v, 8, 32);
  return v;
}

// ---------------------------------------------------------------------------
// Pass 0: build f16 WMMA operand arrays from current h.
//   XA[i][k] = (k==0 ? -x : x)  (Minkowski sign folded into A side), pad->0
//   XB[i][k] = x, pad->0
//   XT[k][i] = x (transposed, rows 17..31 zero) for the B@X B-operand
// ---------------------------------------------------------------------------
__global__ void k_prep(const float* __restrict__ hsrc, const float* __restrict__ theta,
                       _Float16* __restrict__ XA, _Float16* __restrict__ XB,
                       _Float16* __restrict__ XT,
                       float* __restrict__ cs, float* __restrict__ sn,
                       float* __restrict__ ys, float* __restrict__ yc, int first) {
  int i = blockIdx.x * blockDim.x + threadIdx.x;
  if (i >= NN) return;
  ys[i] = 0.0f; yc[i] = 0.0f;
#pragma unroll
  for (int d = 0; d < PADK; ++d) {
    float x = (d < DP1) ? hsrc[i * DP1 + d] : 0.0f;
    XB[i * PADK + d] = (_Float16)x;
    XA[i * PADK + d] = (_Float16)((d == 0) ? -x : x);
    XT[(size_t)d * NN + i] = (_Float16)x;
  }
  if (first) {
    float s, c;
    sincosf(theta[i], &s, &c);
    cs[i] = c; sn[i] = s;
  }
}

// ---------------------------------------------------------------------------
// Pass 1: per 16-row tile, sweep j: G tile by WMMA, then C = adj*exp(-2*dist),
// accumulate ys = C@sin(theta), yc = C@cos(theta).
// ---------------------------------------------------------------------------
__global__ __launch_bounds__(256) void k_pair1(const _Float16* __restrict__ XA,
                                               const _Float16* __restrict__ XB,
                                               const float* __restrict__ adj,
                                               const float* __restrict__ cs,
                                               const float* __restrict__ sn,
                                               float* __restrict__ ys, float* __restrict__ yc) {
  int lane = threadIdx.x & 31;
  int wv   = blockIdx.x * 8 + (threadIdx.x >> 5);   // 256 row tiles
  int i0   = wv * 16;
  int hi   = lane >> 4, col = lane & 15;

  v16h a = load_op16x32(XA + (size_t)i0 * PADK, PADK, lane);
  float ysp[8] = {0.f}, ycp[8] = {0.f};

  for (int j0 = 0; j0 < NN; j0 += 16) {
    v16h b = load_op16x32(XB + (size_t)j0 * PADK, PADK, lane);
    v8f g = {};
    g = __builtin_amdgcn_wmma_f32_16x16x32_f16(false, a, false, b, (short)0, g, false, false);
    int j = j0 + col;
    float snj = sn[j], csj = cs[j];
    if (j0 + 16 < NN)
      __builtin_prefetch(adj + (size_t)(i0 + 8 * hi) * NN + j0 + 16, 0, 1);
#pragma unroll
    for (int r = 0; r < 8; ++r) {
      int row = i0 + r + 8 * hi;
      float G  = g[r];
      float A  = adj[(size_t)row * NN + j];
      float u  = fmaxf(-G, 1.0f);          // clip(-K*G, 1)
      float dm = acoshf(u);                // hyperbolic distance (SQRT_K=1)
      float C  = A * __expf(-2.0f * dm);   // exp(-d/TAU), TAU=0.5
      ysp[r] += C * snj;
      ycp[r] += C * csj;
    }
  }
#pragma unroll
  for (int r = 0; r < 8; ++r) {
    float s1 = half16_reduce(ysp[r]);
    float s2 = half16_reduce(ycp[r]);
    if (col == 0) {
      ys[i0 + r + 8 * hi] = s1;
      yc[i0 + r + 8 * hi] = s2;
    }
  }
}

// ---------------------------------------------------------------------------
// Pass 2 (tiny): phase Euler step, cs2/sn2, order parameter r.
// ---------------------------------------------------------------------------
__global__ __launch_bounds__(1024) void k_theta(const float* __restrict__ theta,
                                                const float* __restrict__ omega,
                                                const float* __restrict__ cs,
                                                const float* __restrict__ sn,
                                                const float* __restrict__ ys,
                                                const float* __restrict__ yc,
                                                float* __restrict__ cs2, float* __restrict__ sn2,
                                                float* __restrict__ outR) {
  __shared__ float rc[1024], rs[1024];
  int t = threadIdx.x;
  float sc = 0.f, ss = 0.f;
  for (int k = 0; k < NN; k += 1024) {
    int i = k + t;
    float dth = omega[i] + (1.0f / (float)NN) * (cs[i] * ys[i] - sn[i] * yc[i]);
    float th  = theta[i] + 0.1f * dth;     // DT_PHASE
    float s2, c2;
    sincosf(th, &s2, &c2);
    cs2[i] = c2; sn2[i] = s2;
    sc += c2; ss += s2;
  }
  rc[t] = sc; rs[t] = ss;
  __syncthreads();
  for (int o = 512; o > 0; o >>= 1) {
    if (t < o) { rc[t] += rc[t + o]; rs[t] += rs[t + o]; }
    __syncthreads();
  }
  if (t == 0) {
    float mc = rc[0] / (float)NN, ms = rs[0] / (float)NN;
    outR[0] = sqrtf(mc * mc + ms * ms);
  }
}

// ---------------------------------------------------------------------------
// Pass 3: sweep j again: G by WMMA, M = C*gate -> d_out, B = M*alpha kept in
// registers, bounced through LDS into WMMA A-operand layout, then m += B @ X
// by two more WMMAs (XT gives the column-contiguous B-operand). Also row-sum
// s_i = sum_j B*G via shuffle reduction. Each wave owns its rows -> no atomics.
// ---------------------------------------------------------------------------
__global__ __launch_bounds__(256) void k_pair2(const _Float16* __restrict__ XA,
                                               const _Float16* __restrict__ XB,
                                               const _Float16* __restrict__ XT,
                                               const float* __restrict__ adj,
                                               const float* __restrict__ cs2,
                                               const float* __restrict__ sn2,
                                               float* __restrict__ Mout,
                                               float* __restrict__ mws, float* __restrict__ sws) {
  __shared__ __align__(16) _Float16 Bl[8][16][40];   // per-wave 16x32 B tile (+pad)
  int lane = threadIdx.x & 31;
  int w    = threadIdx.x >> 5;
  int wv   = blockIdx.x * 8 + w;
  int i0   = wv * 16;
  int hi   = lane >> 4, col = lane & 15;

  v16h a = load_op16x32(XA + (size_t)i0 * PADK, PADK, lane);
  float c2r[8], s2r[8];
#pragma unroll
  for (int r = 0; r < 8; ++r) {
    int row = i0 + r + 8 * hi;
    c2r[r] = cs2[row]; s2r[r] = sn2[row];
  }

  v8f accm0 = {}, accm1 = {};
  float sp[8] = {0.f};

  for (int j0 = 0; j0 < NN; j0 += 32) {
#pragma unroll
    for (int sub = 0; sub < 2; ++sub) {
      int jb = j0 + 16 * sub;
      v16h b = load_op16x32(XB + (size_t)jb * PADK, PADK, lane);
      v8f g = {};
      g = __builtin_amdgcn_wmma_f32_16x16x32_f16(false, a, false, b, (short)0, g, false, false);
      int j = jb + col;
      float c2j = cs2[j], s2j = sn2[j];
#pragma unroll
      for (int r = 0; r < 8; ++r) {
        int row = i0 + r + 8 * hi;
        float G  = g[r];
        float A  = adj[(size_t)row * NN + j];
        float u  = fmaxf(-G, 1.0f);
        float dm = acoshf(u);
        float C  = A * __expf(-2.0f * dm);
        float Mv = C * (0.5f * (1.0f + c2r[r] * c2j + s2r[r] * s2j));
        Mout[(size_t)row * NN + j] = Mv;
        float un = sqrtf(fmaxf(G * G - 1.0f, EPS_F));  // ||logmap dir||_L
        float al = (dm > 1e-6f) ? dm / un : 0.0f;
        float Bv = Mv * al;
        sp[r] += Bv * G;
        Bl[w][r + 8 * hi][col + 16 * sub] = (_Float16)Bv;
      }
    }
    // intra-wave LDS exchange: all lanes' ds_stores must land before re-load
    asm volatile("s_wait_dscnt 0x0" ::: "memory");
    v16h aB  = load_op16x32(&Bl[w][0][0], 40, lane);
    v16h bx0 = load_op16x32(XT + j0, NN, lane);                       // d = 0..15
    v16h bx1 = load_op16x32(XT + (size_t)16 * NN + j0, NN, lane);     // d = 16 (+pad)
    accm0 = __builtin_amdgcn_wmma_f32_16x16x32_f16(false, aB, false, bx0, (short)0, accm0, false, false);
    accm1 = __builtin_amdgcn_wmma_f32_16x16x32_f16(false, aB, false, bx1, (short)0, accm1, false, false);
  }

#pragma unroll
  for (int r = 0; r < 8; ++r) {
    int row = i0 + r + 8 * hi;
    mws[row * DP1 + col] = accm0[r];             // d = col (0..15)
    if (col == 0) mws[row * DP1 + 16] = accm1[r]; // d = 16
    float sv = half16_reduce(sp[r]);
    if (col == 0) sws[row] = sv;
  }
}

// ---------------------------------------------------------------------------
// Pass 4: per-node MLP + tangent projection + exp-map (64 threads / node).
// Safe to run in place on the h buffer.
// ---------------------------------------------------------------------------
__global__ __launch_bounds__(64) void k_update(const float* __restrict__ hsrc,
                                               const float* __restrict__ mws,
                                               const float* __restrict__ sws,
                                               const float* __restrict__ W1,
                                               const float* __restrict__ b1,
                                               const float* __restrict__ W2,
                                               const float* __restrict__ b2,
                                               float* __restrict__ hout) {
  __shared__ float xr[DP1], mr[DP1], hid[64], vv[DP1], sc[2];
  int i = blockIdx.x, t = threadIdx.x;
  if (t < DP1) {
    float x = hsrc[i * DP1 + t];
    xr[t] = x;
    mr[t] = mws[i * DP1 + t] + sws[i] * x;       // m = B@X + (sum B*G) * x
  }
  __syncthreads();
  float acc = b1[t];
#pragma unroll
  for (int d = 0; d < DP1; ++d) acc += mr[d] * W1[d * 64 + t];
  hid[t] = tanhf(acc);
  __syncthreads();
  if (t < DP1) {
    float v = b2[t];
#pragma unroll
    for (int k = 0; k < 64; ++k) v += hid[k] * W2[k * DP1 + t];
    vv[t] = v;
  }
  __syncthreads();
  if (t == 0) {                                   // <x,v>_L
    float md = -xr[0] * vv[0];
    for (int d = 1; d < DP1; ++d) md += xr[d] * vv[d];
    sc[0] = md;
  }
  __syncthreads();
  if (t < DP1) vv[t] = 0.1f * (vv[t] + sc[0] * xr[t]);  // proj, then *DT_NCA
  __syncthreads();
  if (t == 0) {                                   // exp-map coefficients
    float nn = -vv[0] * vv[0];
    for (int d = 1; d < DP1; ++d) nn += vv[d] * vv[d];
    float n = sqrtf(fmaxf(nn, EPS_F));
    sc[0] = coshf(n);
    sc[1] = sinhf(n) / n;
  }
  __syncthreads();
  if (t < DP1) hout[i * DP1 + t] = sc[0] * xr[t] + sc[1] * vv[t];
}

// ---------------------------------------------------------------------------
extern "C" void kernel_launch(void* const* d_in, const int* in_sizes, int n_in,
                              void* d_out, int out_size, void* d_ws, size_t ws_size,
                              hipStream_t stream) {
  const float* adj   = (const float*)d_in[0];
  const float* h0    = (const float*)d_in[1];
  const float* theta = (const float*)d_in[2];
  const float* omega = (const float*)d_in[3];
  const float* W1    = (const float*)d_in[4];
  const float* b1    = (const float*)d_in[5];
  const float* W2    = (const float*)d_in[6];
  const float* b2    = (const float*)d_in[7];
  (void)in_sizes; (void)n_in; (void)out_size; (void)ws_size;

  float* out  = (float*)d_out;
  float* outH = out;                                   // [N,17]
  float* outM = out + (size_t)NN * DP1;                // [N,N]
  float* outR = out + (size_t)NN * DP1 + (size_t)NN * NN;  // scalar

  char* ws = (char*)d_ws;
  size_t off = 0;
  auto take = [&](size_t bytes) { char* p = ws + off; off = (off + bytes + 255) & ~(size_t)255; return p; };
  _Float16* XA = (_Float16*)take((size_t)NN * PADK * 2);
  _Float16* XB = (_Float16*)take((size_t)NN * PADK * 2);
  _Float16* XT = (_Float16*)take((size_t)PADK * NN * 2);
  float* cs  = (float*)take(NN * 4);
  float* sn  = (float*)take(NN * 4);
  float* cs2 = (float*)take(NN * 4);
  float* sn2 = (float*)take(NN * 4);
  float* ys  = (float*)take(NN * 4);
  float* yc  = (float*)take(NN * 4);
  float* sws = (float*)take(NN * 4);
  float* mws = (float*)take((size_t)NN * DP1 * 4);

  const int NUM_STEPS = 2;   // matches reference setup_inputs (num_steps=2)
  for (int step = 0; step < NUM_STEPS; ++step) {
    const float* hsrc = (step == 0) ? h0 : outH;
    k_prep <<<NN / 256, 256, 0, stream>>>(hsrc, theta, XA, XB, XT, cs, sn, ys, yc, step == 0 ? 1 : 0);
    k_pair1<<<32, 256, 0, stream>>>(XA, XB, adj, cs, sn, ys, yc);
    k_theta<<<1, 1024, 0, stream>>>(theta, omega, cs, sn, ys, yc, cs2, sn2, outR);
    k_pair2<<<32, 256, 0, stream>>>(XA, XB, XT, adj, cs2, sn2, outM, mws, sws);
    k_update<<<NN, 64, 0, stream>>>(hsrc, mws, sws, W1, b1, W2, b2, outH);
  }
}
